// BertLayer_22393959481720
// MI455X (gfx1250) — compile-verified, matrix operand order checked by reading, not run on hardware
//
#include <hip/hip_runtime.h>

// ---------------------------------------------------------------------------
// Shapes (fixed by the reference): B=32, S=512, H=768, MS=128, SL=8
//   out_vectors : (32,128,768) f32   -> d_out[0 .. 3145728)
//   sent_repr   : (32,768)     f32   -> d_out[3145728 .. 3170304)
// Workspace: Bp (bf16 packed W_attn, 768*768*2 = 1179648 B), logits (16384 f32)
// ---------------------------------------------------------------------------

typedef __attribute__((ext_vector_type(16))) __bf16 v16bf;
typedef __attribute__((ext_vector_type(8)))  float  v8f;

#define HH   768
#define SS   512
#define BB   32
#define MSEG 128
#define SLEN 8
#define KT   (HH / 32)   // 24 K-steps of 32
#define NT   (HH / 16)   // 48 N-tiles of 16

// Branch-free tanh: native v_tanh_f32 if the builtin exists, else one v_exp_f32.
#if __has_builtin(__builtin_amdgcn_tanhf)
__device__ __forceinline__ float fast_tanh(float x) {
    return __builtin_amdgcn_tanhf(x);
}
#elif __has_builtin(__builtin_amdgcn_tanh_f32)
__device__ __forceinline__ float fast_tanh(float x) {
    return __builtin_amdgcn_tanh_f32(x);
}
#else
__device__ __forceinline__ float fast_tanh(float x) {
    // tanh(x) = 1 - 2/(exp(2x)+1); saturates correctly for |x| large.
    float e = __expf(2.0f * x);
    return 1.0f - 2.0f / (e + 1.0f);
}
#endif

// ---------------------------------------------------------------------------
// Pack W_attn (f32, row-major [k][n]) into bf16 WMMA B-fragment layout:
//   Bp[((kt*NT + nt)*32 + lane)*16 + e]
//   k = kt*32 + base0 + (e&7) + (e>=8 ? 16 : 0),  base0 = (lane>=16) ? 8 : 0
//   n = nt*16 + (lane&15)
// ---------------------------------------------------------------------------
__global__ void __launch_bounds__(256) pack_w(const float* __restrict__ W,
                                              __bf16* __restrict__ Bp) {
    int o = blockIdx.x * 256 + threadIdx.x;          // 0 .. 768*768
    int e    = o & 15;
    int lane = (o >> 4) & 31;
    int frag = o >> 9;                               // kt*NT + nt
    int nt = frag % NT;
    int kt = frag / NT;
    int base0 = (lane >= 16) ? 8 : 0;
    int k = kt * 32 + base0 + (e & 7) + ((e >= 8) ? 16 : 0);
    int n = nt * 16 + (lane & 15);
    Bp[o] = (__bf16)W[k * HH + n];
}

// ---------------------------------------------------------------------------
// Fused GEMM + tanh + dot(w_ctx): logits[row] = sum_n tanh((X*W)[row,n]+b[n])*wc[n]
// Block = 256 threads (8 waves), computes 128 rows. A tile staged once in LDS
// (bf16, fragment-swizzled, 128*768*2 = 192 KB dynamic LDS). N-loop unrolled
// by 2: each LDS A-fragment feeds two WMMAs (two accumulators).
// ---------------------------------------------------------------------------
__global__ void __launch_bounds__(256)
gemm_logits(const float*  __restrict__ x,
            const __bf16* __restrict__ Bp,
            const float*  __restrict__ b_attn,
            const float*  __restrict__ w_ctx,
            float*        __restrict__ logits) {
    extern __shared__ __bf16 sA[];                   // [rg][kt][lane][16]
    const int  tid  = threadIdx.x;
    const int  lane = tid & 31;
    const int  wv   = tid >> 5;                      // wave id = row group
    const long rowbase = (long)blockIdx.x * 128;

    // ---- stage A tile (f32 -> bf16, swizzled to A-fragment layout) ----
    for (int r = 0; r < 128; ++r) {
        const float* xr = x + (rowbase + r) * HH;
        int rg = r >> 4;
        int ml = r & 15;
        for (int k = tid; k < HH; k += 256) {        // coalesced global reads
            int kt = k >> 5;
            int kl = k & 31;
            int ls = ml + ((kl & 8) ? 16 : 0);       // lane slot
            int e  = (kl & 7) + ((kl >= 16) ? 8 : 0);
            sA[(((rg * KT + kt) * 32 + ls) << 4) + e] = (__bf16)xr[k];
        }
    }
    __syncthreads();

    // ---- WMMA: 16 rows per wave, N-tiles in pairs, inner K loop ----
    const __bf16* aw = sA + (size_t)wv * KT * 512;   // (rg*KT+kt)*32*16 elems
    const int ncol = lane & 15;

    float lacc[8];
#pragma unroll
    for (int v = 0; v < 8; ++v) lacc[v] = 0.0f;

    for (int nt = 0; nt < NT; nt += 2) {
        v8f c0 = {};
        v8f c1 = {};
        const __bf16* bp0 = Bp + ((size_t)(nt * 32 + lane) << 4);
        const __bf16* bp1 = bp0 + (32 << 4);         // next N-tile
#pragma unroll 4
        for (int kt = 0; kt < KT; ++kt) {
            v16bf a  = *(const v16bf*)(aw + kt * 512 + (lane << 4));
            v16bf b0 = *(const v16bf*)(bp0 + (size_t)kt * (NT * 512));
            v16bf b1 = *(const v16bf*)(bp1 + (size_t)kt * (NT * 512));
            c0 = __builtin_amdgcn_wmma_f32_16x16x32_bf16(
                     false, a, false, b0, (short)0, c0, false, false);
            c1 = __builtin_amdgcn_wmma_f32_16x16x32_bf16(
                     false, a, false, b1, (short)0, c1, false, false);
        }
        int   n0  = nt * 16 + ncol;
        int   n1  = n0 + 16;
        float ba0 = b_attn[n0], wc0 = w_ctx[n0];
        float ba1 = b_attn[n1], wc1 = w_ctx[n1];
#pragma unroll
        for (int v = 0; v < 8; ++v) {                // C layout: n=lane&15, m=v+8*(lane>=16)
            lacc[v] += fast_tanh(c0[v] + ba0) * wc0;
            lacc[v] += fast_tanh(c1[v] + ba1) * wc1;
        }
    }

    // reduce the 16 columns held across each 16-lane half
#pragma unroll
    for (int v = 0; v < 8; ++v) {
        float s = lacc[v];
        s += __shfl_xor(s, 1, 32);
        s += __shfl_xor(s, 2, 32);
        s += __shfl_xor(s, 4, 32);
        s += __shfl_xor(s, 8, 32);
        lacc[v] = s;
    }
    if ((lane & 15) == 0) {
        long row0 = rowbase + wv * 16 + ((lane >> 4) << 3);  // +0 or +8
#pragma unroll
        for (int v = 0; v < 8; ++v) logits[row0 + v] = lacc[v];
    }
}

// ---------------------------------------------------------------------------
// Softmax over S=512 logits per batch, then sent_repr[b,h] = sum_s a[s]*x[b,s,h]
// ---------------------------------------------------------------------------
__global__ void __launch_bounds__(256)
softmax_repr(const float* __restrict__ x,
             const float* __restrict__ logits,
             float* __restrict__ sent) {
    __shared__ float w[SS];
    __shared__ float red[256];
    const int b   = blockIdx.x;
    const int tid = threadIdx.x;

    float l0 = logits[b * SS + tid];
    float l1 = logits[b * SS + 256 + tid];

    red[tid] = fmaxf(l0, l1);
    __syncthreads();
    for (int s2 = 128; s2 > 0; s2 >>= 1) {
        if (tid < s2) red[tid] = fmaxf(red[tid], red[tid + s2]);
        __syncthreads();
    }
    float mx = red[0];
    __syncthreads();

    float e0 = __expf(l0 - mx), e1 = __expf(l1 - mx);
    red[tid] = e0 + e1;
    __syncthreads();
    for (int s2 = 128; s2 > 0; s2 >>= 1) {
        if (tid < s2) red[tid] += red[tid + s2];
        __syncthreads();
    }
    float inv = 1.0f / red[0];
    w[tid]       = e0 * inv;
    w[tid + 256] = e1 * inv;
    __syncthreads();

    for (int h = tid; h < HH; h += 256) {
        const float* xb = x + (long)b * SS * HH + h;
        float acc = 0.0f;
        for (int s = 0; s < SS; ++s) acc += w[s] * xb[(long)s * HH];
        sent[b * HH + h] = acc;
    }
}

// ---------------------------------------------------------------------------
// Segment masked max-pool with pad_vec = x[b, S-2, :] fallback.
// One block per (b, seg); 256 threads x 3 h-columns.
// ---------------------------------------------------------------------------
__global__ void __launch_bounds__(256)
segmax(const float* __restrict__ x,
       const int*   __restrict__ segments,
       const unsigned char* __restrict__ seg_mask,
       const unsigned char* __restrict__ idx_mask,
       float* __restrict__ out_vec) {
    const int bs  = blockIdx.x;                  // b*MSEG + seg
    const int b   = bs >> 7;
    const int tid = threadIdx.x;
    const bool sm = seg_mask[bs] != 0;

    int   sidx[SLEN];
    bool  im[SLEN];
#pragma unroll
    for (int j = 0; j < SLEN; ++j) {
        sidx[j] = segments[bs * SLEN + j];
        im[j]   = idx_mask[bs * SLEN + j] != 0;
    }

    const float* xb  = x + (long)b * SS * HH;
    const float* pad = xb + (long)(SS - 2) * HH;

#pragma unroll
    for (int t = 0; t < 3; ++t) {
        int   h = tid + t * 256;
        float m = -1e30f;
#pragma unroll
        for (int j = 0; j < SLEN; ++j)
            if (im[j]) m = fmaxf(m, xb[(long)sidx[j] * HH + h]);
        if (!sm) m = pad[h];
        out_vec[(long)bs * HH + h] = m;
    }
}

// ---------------------------------------------------------------------------
extern "C" void kernel_launch(void* const* d_in, const int* in_sizes, int n_in,
                              void* d_out, int out_size, void* d_ws, size_t ws_size,
                              hipStream_t stream) {
    const float*         x        = (const float*)d_in[0];          // (32,512,768)
    const int*           segments = (const int*)d_in[1];            // (32,128,8)
    const unsigned char* smask    = (const unsigned char*)d_in[2];  // (32,128) bool
    const unsigned char* simask   = (const unsigned char*)d_in[3];  // (32,128,8) bool
    const float*         W        = (const float*)d_in[4];          // (768,768)
    const float*         b_attn   = (const float*)d_in[5];          // (768,)
    const float*         w_ctx    = (const float*)d_in[6];          // (768,1)

    float* out_vec = (float*)d_out;                                 // 32*128*768
    float* sent    = out_vec + (size_t)BB * MSEG * HH;              // 32*768

    __bf16* Bp     = (__bf16*)d_ws;                                 // 768*768 bf16
    float*  logits = (float*)((char*)d_ws + (size_t)HH * HH * 2);   // 16384 f32

    // 1) pack W_attn to bf16 WMMA fragments
    pack_w<<<(HH * HH) / 256, 256, 0, stream>>>(W, Bp);

    // 2) fused GEMM(bf16 WMMA, f32 acc) + tanh + dot(w_ctx) -> logits
    gemm_logits<<<(BB * SS) / 128, 256, 128 * HH * 2, stream>>>(
        x, Bp, b_attn, w_ctx, logits);

    // 3) softmax over S + weighted sum -> sent_repr
    softmax_repr<<<BB, 256, 0, stream>>>(x, logits, sent);

    // 4) segment masked max-pool -> out_vectors
    segmax<<<BB * MSEG, 256, 0, stream>>>(x, segments, smask, simask, out_vec);
}
